// FlaxBloomAttention_62801011802243
// MI455X (gfx1250) — matrix operand order repaired and checked by reading, not compiled, not run on hardware
//
#include <hip/hip_runtime.h>
#include <hip/hip_bf16.h>

typedef __attribute__((ext_vector_type(16))) __bf16 v16bf;
typedef __attribute__((ext_vector_type(8)))  __bf16 v8bf;
typedef __attribute__((ext_vector_type(8)))  float  v8f;

// ---------------------------------------------------------------------------
// Problem constants (B=2, S=2048, H=1024, 16 heads, hd=64)
// ---------------------------------------------------------------------------
#define BATCH   2
#define SEQ     2048
#define HID     1024
#define NHEAD   16
#define HD      64
#define ROWS    (BATCH * SEQ)      // 4096
#define QKVN    (3 * HID)          // 3072

// ---------------------------------------------------------------------------
// WMMA helpers: V_WMMA_F32_16X16X32_BF16, D = A(16x32) * B(32x16) + C
// A layout: lane m=L&15, half=L>>4; slots 0..7 -> K=half*8+j ; slots 8..15 -> K=16+half*8+j
// B layout: lane n=L&15, half=L>>4; slots 0..7 -> K=half*8+j ; slots 8..15 -> K=16+half*8+j
// C layout: lane n=L&15; vgpr r -> row m = r + 8*(L>>4)
// ---------------------------------------------------------------------------
__device__ __forceinline__ v8f wmma_bf16(v16bf a, v16bf b, v8f c) {
  return __builtin_amdgcn_wmma_f32_16x16x32_bf16(false, a, false, b, (short)0, c,
                                                 false, false);
}

// A fragment from a row-major (M x K) tile, rows at row_base, leading dim ld.
__device__ __forceinline__ v16bf load_a_frag(const __bf16* row_base, int ld,
                                             int k0, int lane) {
  const int m = lane & 15, half = lane >> 4;
  const __bf16* p = row_base + (size_t)m * ld + k0 + half * 8;
  union { v16bf v; v8bf h[2]; } u;
  u.h[0] = *(const v8bf*)p;           // K = k0 + half*8 + [0..7]
  u.h[1] = *(const v8bf*)(p + 16);    // K = k0 + 16 + half*8 + [0..7]
  return u.v;
}

// B fragment from an n-major (N x K) tile (i.e. B^T row-major), leading dim ld.
__device__ __forceinline__ v16bf load_b_frag(const __bf16* nrow_base, int ld,
                                             int k0, int lane) {
  const int n = lane & 15, half = lane >> 4;
  union { v16bf v; v8bf h[2]; } u;
  const __bf16* p = nrow_base + (size_t)n * ld + k0 + half * 8;
  u.h[0] = *(const v8bf*)p;           // K = k0 + half*8 + [0..7]
  u.h[1] = *(const v8bf*)(p + 16);    // K = k0 + 16 + half*8 + [0..7]
  return u.v;
}

// ---------------------------------------------------------------------------
// Conversion kernels
// ---------------------------------------------------------------------------
__global__ void cvt_f32_bf16(const float* __restrict__ in, __bf16* __restrict__ out,
                             int n) {
  int i = blockIdx.x * blockDim.x + threadIdx.x;
  if (i < n) out[i] = (__bf16)in[i];
}

// in: (rows x cols) row-major f32 -> out: (cols x rows) row-major bf16
__global__ void transpose_cvt(const float* __restrict__ in, __bf16* __restrict__ out,
                              int rows, int cols) {
  int i = blockIdx.x * blockDim.x + threadIdx.x;
  if (i < rows * cols) {
    int r = i / cols, c = i % cols;
    out[(size_t)c * rows + r] = (__bf16)in[i];
  }
}

// ---------------------------------------------------------------------------
// Kernel 1: fused QKV GEMM  C(4096 x 3072) = X(4096x1024) @ Wqkv + b
// Scatters into Q (b,h,s,d), K (b,h,s,d), Vt (b,h,d,s) as bf16.
// Block tile 128x128 from 4 waves (2x2); wave tile 64x64 = 4x4 WMMA accums.
// Per k-step: 16 WMMAs on 4 A-frags + 4 B-frags. Branch-free loop body,
// fully unrolled epilogue (accumulators must stay in VGPRs, no scratch).
// ---------------------------------------------------------------------------
__global__ __launch_bounds__(128) void qkv_gemm(
    const __bf16* __restrict__ X, const __bf16* __restrict__ Wt,
    const float* __restrict__ bias, __bf16* __restrict__ Q,
    __bf16* __restrict__ K, __bf16* __restrict__ Vt) {
  const int lane = threadIdx.x & 31;
  const int wave = threadIdx.x >> 5;            // 0..3
  const int m0 = blockIdx.x * 128 + (wave >> 1) * 64;
  const int n0 = blockIdx.y * 128 + (wave & 1) * 64;

  const __bf16* A0 = X + (size_t)m0 * HID;
  const __bf16* B0 = Wt + (size_t)n0 * HID;

  v8f acc[4][4] = {};
  for (int k0 = 0; k0 < HID; k0 += 32) {
    // speculative prefetch of the next k-slab (dropped silently if stale)
    __builtin_prefetch(A0 + (size_t)(lane & 15) * HID + k0 + 32, 0, 1);
    __builtin_prefetch(B0 + (size_t)(lane & 15) * HID + k0 + 32, 0, 1);
    v16bf a[4], b[4];
#pragma unroll
    for (int i = 0; i < 4; ++i)
      a[i] = load_a_frag(A0 + (size_t)(16 * i) * HID, HID, k0, lane);
#pragma unroll
    for (int j = 0; j < 4; ++j)
      b[j] = load_b_frag(B0 + (size_t)(16 * j) * HID, HID, k0, lane);
#pragma unroll
    for (int i = 0; i < 4; ++i)
#pragma unroll
      for (int j = 0; j < 4; ++j)
        acc[i][j] = wmma_bf16(a[i], b[j], acc[i][j]);
  }

  const int nl = lane & 15, half = lane >> 4;
#pragma unroll
  for (int i = 0; i < 4; ++i)
#pragma unroll
    for (int j = 0; j < 4; ++j) {
      const int col = n0 + j * 16 + nl;       // 0..3071
      const int head = col / (3 * HD);
      const int idx = col % (3 * HD);
      const float bv = bias[col];
#pragma unroll
      for (int r = 0; r < 8; ++r) {
        const int row = m0 + i * 16 + r + 8 * half;  // 0..4095
        const int bi = row >> 11;                    // batch
        const int s = row & (SEQ - 1);
        const float v = acc[i][j][r] + bv;
        const __bf16 bb = (__bf16)v;
        const size_t bh = (size_t)(bi * NHEAD + head);
        if (idx < HD)
          Q[(bh * SEQ + s) * HD + idx] = bb;
        else if (idx < 2 * HD)
          K[(bh * SEQ + s) * HD + (idx - HD)] = bb;
        else
          Vt[(bh * HD + (idx - 2 * HD)) * SEQ + s] = bb;
      }
    }
}

// ---------------------------------------------------------------------------
// Kernel 2: causal flash attention.
// Block = 4 waves, each wave owns 16 q-rows (block covers 64). Iterate 128
// k-columns per step: 8 score tiles (2 chained WMMAs each over hd=64), online
// softmax via shfl_xor row reductions, P staged bf16 in LDS, 16 context WMMAs.
// scores = qk/8 + alibi (layer scaling cancels), mask -> -10000.
// ---------------------------------------------------------------------------
__global__ __launch_bounds__(128) void flash_attn(
    const __bf16* __restrict__ Q, const __bf16* __restrict__ K,
    const __bf16* __restrict__ Vt, const float* __restrict__ alibi,
    const unsigned char* __restrict__ amask, __bf16* __restrict__ Ctx) {
  __shared__ __bf16 Plds[4][16][136];   // per-wave 16x128 P tile, padded stride

  const int lane = threadIdx.x & 31;
  const int wave = threadIdx.x >> 5;
  const int bh = blockIdx.y;            // b*NHEAD + h
  const int b = bh >> 4, h = bh & 15;
  const int q0 = blockIdx.x * 64 + wave * 16;

  const __bf16* Qb = Q + ((size_t)bh * SEQ + q0) * HD;
  const __bf16* Kb = K + (size_t)bh * SEQ * HD;
  const __bf16* Vb = Vt + (size_t)bh * HD * SEQ;
  const float* al = alibi + (size_t)bh * SEQ;
  const unsigned char* pad = amask + (size_t)b * SEQ;

  const int nl = lane & 15, half = lane >> 4;

  // Q fragments for the whole hd=64 (two K-steps), kept in registers.
  v16bf qa0 = load_a_frag(Qb, HD, 0, lane);
  v16bf qa1 = load_a_frag(Qb, HD, 32, lane);

  bool rowok[8];
#pragma unroll
  for (int r = 0; r < 8; ++r) rowok[r] = pad[q0 + r + 8 * half] != 0;

  v8f ctx[4] = {};
  float rmax[8], rsum[8];
#pragma unroll
  for (int r = 0; r < 8; ++r) { rmax[r] = -1e30f; rsum[r] = 0.0f; }

  const int kend = q0 + 15;             // causal upper bound for this wave
  for (int kb = 0; kb <= kend; kb += 128) {
    v8f sc[8];
    // ---- scores for 8 x (16x16) tiles --------------------------------
#pragma unroll
    for (int j = 0; j < 8; ++j) {
      const __bf16* krow = Kb + (size_t)(kb + j * 16) * HD;
      v16bf kf0 = load_b_frag(krow, HD, 0, lane);
      v16bf kf1 = load_b_frag(krow, HD, 32, lane);
      v8f s = {};
      s = wmma_bf16(qa0, kf0, s);
      s = wmma_bf16(qa1, kf1, s);
      const int kcol = kb + j * 16 + nl;
      const float av = al[kcol];
      const bool kok = pad[kcol] != 0;
#pragma unroll
      for (int r = 0; r < 8; ++r) {
        const int qrow = q0 + r + 8 * half;
        const float v = s[r] * 0.125f + av;       // 1/sqrt(64)
        const bool keep = (kcol <= qrow) && kok && rowok[r];
        s[r] = keep ? v : -10000.0f;
      }
      sc[j] = s;
    }
    // ---- online softmax: block row-max, rescale ----------------------
    float scale[8];
#pragma unroll
    for (int r = 0; r < 8; ++r) {
      float m = sc[0][r];
#pragma unroll
      for (int j = 1; j < 8; ++j) m = fmaxf(m, sc[j][r]);
#pragma unroll
      for (int off = 1; off < 16; off <<= 1) m = fmaxf(m, __shfl_xor(m, off, 32));
      const float nm = fmaxf(rmax[r], m);
      scale[r] = __expf(rmax[r] - nm);
      rmax[r] = nm;
    }
    float bsum[8] = {0, 0, 0, 0, 0, 0, 0, 0};
#pragma unroll
    for (int j = 0; j < 8; ++j)
#pragma unroll
      for (int r = 0; r < 8; ++r) {
        const float p = __expf(sc[j][r] - rmax[r]);
        bsum[r] += p;
        Plds[wave][r + 8 * half][j * 16 + nl] = (__bf16)p;
      }
#pragma unroll
    for (int r = 0; r < 8; ++r) {
      float s = bsum[r];
#pragma unroll
      for (int off = 1; off < 16; off <<= 1) s += __shfl_xor(s, off, 32);
      rsum[r] = rsum[r] * scale[r] + s;
    }
#pragma unroll
    for (int t = 0; t < 4; ++t)
#pragma unroll
      for (int r = 0; r < 8; ++r) ctx[t][r] *= scale[r];

    asm volatile("s_wait_dscnt 0" ::: "memory");  // P tile visible to own wave

    // ---- context: P(16x128) @ V(128x64), 4 K-steps x 4 n-tiles -------
#pragma unroll
    for (int ks = 0; ks < 4; ++ks) {
      v16bf pa = load_a_frag(&Plds[wave][0][0], 136, ks * 32, lane);
#pragma unroll
      for (int t = 0; t < 4; ++t) {
        v16bf vb = load_b_frag(Vb + (size_t)(t * 16) * SEQ, SEQ, kb + ks * 32, lane);
        ctx[t] = wmma_bf16(pa, vb, ctx[t]);
      }
    }
  }

  // ---- normalize and emit context (bf16, (B*S, H) row-major) ---------
  float inv[8];
#pragma unroll
  for (int r = 0; r < 8; ++r)
    inv[r] = rowok[r] ? (1.0f / rsum[r]) : 0.0f;   // probs * keep for padded rows
  const int row_base = b * SEQ + q0;
#pragma unroll
  for (int t = 0; t < 4; ++t)
#pragma unroll
    for (int r = 0; r < 8; ++r) {
      const float v = ctx[t][r] * inv[r];
      const int row = row_base + r + 8 * half;
      const int col = h * HD + t * 16 + nl;
      Ctx[(size_t)row * HID + col] = (__bf16)v;
    }
}

// ---------------------------------------------------------------------------
// Kernel 3: output projection  out = Ctx(4096x1024) @ Wo + b_o + residual (f32)
// Same 128x128 block / 64x64 wave tiling as qkv_gemm.
// ---------------------------------------------------------------------------
__global__ __launch_bounds__(128) void out_proj(
    const __bf16* __restrict__ Ctx, const __bf16* __restrict__ WoT,
    const float* __restrict__ bias, const float* __restrict__ residual,
    float* __restrict__ out) {
  const int lane = threadIdx.x & 31;
  const int wave = threadIdx.x >> 5;            // 0..3
  const int m0 = blockIdx.x * 128 + (wave >> 1) * 64;
  const int n0 = blockIdx.y * 128 + (wave & 1) * 64;

  const __bf16* A0 = Ctx + (size_t)m0 * HID;
  const __bf16* B0 = WoT + (size_t)n0 * HID;

  v8f acc[4][4] = {};
  for (int k0 = 0; k0 < HID; k0 += 32) {
    __builtin_prefetch(A0 + (size_t)(lane & 15) * HID + k0 + 32, 0, 1);
    __builtin_prefetch(B0 + (size_t)(lane & 15) * HID + k0 + 32, 0, 1);
    v16bf a[4], b[4];
#pragma unroll
    for (int i = 0; i < 4; ++i)
      a[i] = load_a_frag(A0 + (size_t)(16 * i) * HID, HID, k0, lane);
#pragma unroll
    for (int j = 0; j < 4; ++j)
      b[j] = load_b_frag(B0 + (size_t)(16 * j) * HID, HID, k0, lane);
#pragma unroll
    for (int i = 0; i < 4; ++i)
#pragma unroll
      for (int j = 0; j < 4; ++j)
        acc[i][j] = wmma_bf16(a[i], b[j], acc[i][j]);
  }

  const int nl = lane & 15, half = lane >> 4;
#pragma unroll
  for (int i = 0; i < 4; ++i)
#pragma unroll
    for (int j = 0; j < 4; ++j) {
      const int col = n0 + j * 16 + nl;
      const float bv = bias[col];
#pragma unroll
      for (int r = 0; r < 8; ++r) {
        const int row = m0 + i * 16 + r + 8 * half;
        const size_t o = (size_t)row * HID + col;
        out[o] = acc[i][j][r] + bv + residual[o];
      }
    }
}

// ---------------------------------------------------------------------------
// Host-side launcher
// ---------------------------------------------------------------------------
extern "C" void kernel_launch(void* const* d_in, const int* in_sizes, int n_in,
                              void* d_out, int out_size, void* d_ws, size_t ws_size,
                              hipStream_t stream) {
  (void)in_sizes; (void)n_in; (void)out_size; (void)ws_size;
  const float* hidden   = (const float*)d_in[0];           // (B,S,H)
  const float* residual = (const float*)d_in[1];           // (B,S,H)
  const float* alibi    = (const float*)d_in[2];           // (B*NH,1,S)
  const unsigned char* amask = (const unsigned char*)d_in[3]; // (B,S) bool
  const float* w_qkv    = (const float*)d_in[4];           // (H,3H)
  const float* b_qkv    = (const float*)d_in[5];           // (3H)
  const float* w_o      = (const float*)d_in[6];           // (H,H)
  const float* b_o      = (const float*)d_in[7];           // (H)
  float* out = (float*)d_out;

  // Workspace layout (bf16 intermediates), ~48 MiB total.
  char* w = (char*)d_ws;
  __bf16* Xbf   = (__bf16*)w;  w += (size_t)ROWS * HID * 2;    // 8 MiB
  __bf16* WqkvT = (__bf16*)w;  w += (size_t)QKVN * HID * 2;    // 6 MiB
  __bf16* WoT   = (__bf16*)w;  w += (size_t)HID * HID * 2;     // 2 MiB
  __bf16* Qb    = (__bf16*)w;  w += (size_t)ROWS * HID * 2;    // 8 MiB
  __bf16* Kb    = (__bf16*)w;  w += (size_t)ROWS * HID * 2;    // 8 MiB
  __bf16* Vt    = (__bf16*)w;  w += (size_t)ROWS * HID * 2;    // 8 MiB
  __bf16* Ctx   = (__bf16*)w;  w += (size_t)ROWS * HID * 2;    // 8 MiB

  // 0) precision conversion / weight transpose
  {
    int n = ROWS * HID;
    cvt_f32_bf16<<<(n + 255) / 256, 256, 0, stream>>>(hidden, Xbf, n);
  }
  {
    int n = HID * QKVN;
    transpose_cvt<<<(n + 255) / 256, 256, 0, stream>>>(w_qkv, WqkvT, HID, QKVN);
  }
  {
    int n = HID * HID;
    transpose_cvt<<<(n + 255) / 256, 256, 0, stream>>>(w_o, WoT, HID, HID);
  }

  // 1) fused QKV projection (WMMA, 64x64 wave tiles)
  qkv_gemm<<<dim3(ROWS / 128, QKVN / 128), 128, 0, stream>>>(Xbf, WqkvT, b_qkv,
                                                             Qb, Kb, Vt);

  // 2) causal flash attention (WMMA + LDS + shfl softmax)
  flash_attn<<<dim3(SEQ / 64, BATCH * NHEAD), 128, 0, stream>>>(Qb, Kb, Vt, alibi,
                                                                amask, Ctx);

  // 3) output projection + bias + residual (WMMA, f32 epilogue)
  out_proj<<<dim3(ROWS / 128, HID / 128), 128, 0, stream>>>(Ctx, WoT, b_o,
                                                            residual, out);
}
